// GraphAttentionLayer_73641509257729
// MI455X (gfx1250) — matrix operand order repaired and checked by reading, not compiled, not run on hardware
//
#include <hip/hip_runtime.h>
#include <math.h>

#define N_NODES 50000
#define N_EDGES 800000
#define IN_F    128
#define HF      128   // H * F (output features)
#define NH      4     // heads
#define F       32    // features per head

typedef __attribute__((ext_vector_type(16))) _Float16 v16h;
typedef __attribute__((ext_vector_type(8)))  _Float16 v8h;
typedef __attribute__((ext_vector_type(8)))  float    v8f;

// ------------------------------------------------------------------
// Kernel 1: zero the atomic accumulation buffers (out + denom).
// ------------------------------------------------------------------
__global__ void gat_zero_kernel(float* __restrict__ out, float* __restrict__ denom) {
    int i = blockIdx.x * blockDim.x + threadIdx.x;
    if (i < N_NODES * HF) out[i] = 0.0f;
    if (i < N_NODES * NH) denom[i] = 0.0f;
}

// ------------------------------------------------------------------
// Kernel 2: Wh = x @ W  via v_wmma_f32_16x16x32_f16 (f32 accumulate).
// One 16-node row tile per block; 8 waves cover the 128 output cols.
// W is staged transposed (f16) in LDS so both fragments load as
// contiguous 16B ds reads. No divergence -> EXEC all-1s for WMMA.
// ------------------------------------------------------------------
__global__ __launch_bounds__(256) void gat_gemm_kernel(const float* __restrict__ x,
                                                       const float* __restrict__ W,
                                                       float* __restrict__ Wh) {
    __shared__ _Float16 sA[16 * 128];    // x tile, [m][k]
    __shared__ _Float16 sBt[128 * 128];  // W transposed, [n][k]  (32 KB)

    const int tid  = threadIdx.x;
    const int row0 = blockIdx.x * 16;    // 3125 * 16 == 50000 exactly

    // stage x tile: 16x128 f32 -> f16  (coalesced)
    #pragma unroll
    for (int r = 0; r < 8; ++r) {
        int idx = tid + r * 256;                 // 0..2047
        sA[idx] = (_Float16)x[row0 * IN_F + idx];
    }
    // stage W: 128x128 f32 -> f16, transposed into [n][k]
    #pragma unroll
    for (int r = 0; r < 64; ++r) {
        int idx = tid + r * 256;                 // 0..16383
        int k = idx >> 7, n = idx & 127;
        sBt[n * 128 + k] = (_Float16)W[idx];
    }
    __syncthreads();

    const int lane = tid & 31;
    const int wave = tid >> 5;                   // 8 waves -> col tiles of 16
    const int n0   = wave * 16;
    const int mn   = lane & 15;                  // A row / B col within tile
    // A-fragment K mapping (ISA 7.12.2, 16-bit A 16x32):
    //   lanes 0-15: elems 0..7 -> K 0..7,  8..15 -> K 16..23
    //   lanes 16-31: +8 on both halves
    const int ka   = (lane & 16) ? 8 : 0;
    // B-fragment K mapping (same as A):
    //   lanes 0-15: elems 0..7 -> K 0..7, 8..15 -> K 16..23; lanes 16-31: +8
    const int kb_off = (lane & 16) ? 8 : 0;

    v8f c = {};
    #pragma unroll
    for (int kb = 0; kb < 4; ++kb) {             // K = 128 in steps of 32
        const int kA = kb * 32 + ka;
        const int kB = kb * 32 + kb_off;
        v8h alo = *(const v8h*)&sA[mn * 128 + kA];
        v8h ahi = *(const v8h*)&sA[mn * 128 + kA + 16];
        v8h blo = *(const v8h*)&sBt[(n0 + mn) * 128 + kB];
        v8h bhi = *(const v8h*)&sBt[(n0 + mn) * 128 + kB + 16];
        v16h a, b;
        #pragma unroll
        for (int i = 0; i < 8; ++i) {
            a[i] = alo[i]; a[i + 8] = ahi[i];
            b[i] = blo[i]; b[i + 8] = bhi[i];
        }
        // (neg_a, A, neg_b, B, c_mod, C, reuse_a, reuse_b)
        c = __builtin_amdgcn_wmma_f32_16x16x32_f16(false, a, false, b, (short)0, c,
                                                   false, false);
    }

    // D layout: VGPR r -> M = r (lanes 0-15) / r+8 (lanes 16-31), N = lane&15
    const int mrow = (lane & 16) ? 8 : 0;
    #pragma unroll
    for (int r = 0; r < 8; ++r)
        Wh[(row0 + r + mrow) * HF + n0 + mn] = c[r];
}

// ------------------------------------------------------------------
// Kernel 3: per-node attention scalars
//   alpha_src[n,h] = <Wh[n,h,:], a[h,0:F]>,  alpha_dst[n,h] = <Wh[n,h,:], a[h,F:2F]>
// Collapses the per-edge work from 2x128-float gathers to 2x4-float gathers.
// ------------------------------------------------------------------
__global__ void gat_alpha_kernel(const float* __restrict__ Wh, const float* __restrict__ a,
                                 float* __restrict__ as, float* __restrict__ ad) {
    int t = blockIdx.x * blockDim.x + threadIdx.x;
    if (t >= N_NODES * NH) return;
    int n = t >> 2, h = t & 3;
    const float* wr = Wh + n * HF + h * F;
    const float* al = a + h * 2 * F;
    float s = 0.f, d = 0.f;
    #pragma unroll
    for (int f = 0; f < F; f += 4) {
        float4 w  = *(const float4*)(wr + f);
        float4 c1 = *(const float4*)(al + f);
        float4 c2 = *(const float4*)(al + F + f);
        s += w.x * c1.x + w.y * c1.y + w.z * c1.z + w.w * c1.w;
        d += w.x * c2.x + w.y * c2.y + w.z * c2.z + w.w * c2.w;
    }
    as[t] = s;
    ad[t] = d;
}

// ------------------------------------------------------------------
// Kernel 4: edge pass 1 — e -> leaky_relu -> exp, accumulate denom[dst].
// alpha tables (800 KB) + denom (800 KB) are L2-resident.
// ------------------------------------------------------------------
__global__ void gat_edge1_kernel(const int* __restrict__ src, const int* __restrict__ dst,
                                 const float* __restrict__ as, const float* __restrict__ ad,
                                 float* __restrict__ expe, float* __restrict__ denom) {
    int e = blockIdx.x * blockDim.x + threadIdx.x;   // 800000 = 3125*256 exactly
    int s = src[e], d = dst[e];
    float4 va = *(const float4*)(as + s * NH);
    float4 vb = *(const float4*)(ad + d * NH);
    float v[4] = {va.x + vb.x, va.y + vb.y, va.z + vb.z, va.w + vb.w};
    #pragma unroll
    for (int h = 0; h < NH; ++h) {
        float t = v[h];
        t = (t > 0.f) ? t : 0.2f * t;                // leaky_relu(0.2)
        float ex = expf(t);
        expe[e * NH + h] = ex;
        atomicAdd(&denom[d * NH + h], ex);
    }
}

// ------------------------------------------------------------------
// Kernel 5: edge pass 2 — scatter att * Wh[src] into out[dst].
// 32 threads per edge: (head, 4-float group). Wh (25.6 MB) and out
// (25.6 MB) both fit the 192 MB L2, so gathers+atomics stay on-chip.
// ------------------------------------------------------------------
__global__ void gat_edge2_kernel(const int* __restrict__ src, const int* __restrict__ dst,
                                 const float* __restrict__ Wh, const float* __restrict__ expe,
                                 const float* __restrict__ denom, float* __restrict__ out) {
    int t  = blockIdx.x * blockDim.x + threadIdx.x;  // E*32 = 100000*256 exactly
    int e  = t >> 5;
    int j  = t & 31;
    int h  = j >> 3;                                 // head 0..3
    int f4 = (j & 7) << 2;                           // feature group base 0..28
    int s = src[e], d = dst[e];
    float att = expe[e * NH + h] / (denom[d * NH + h] + 1e-10f);
    float4 w = *(const float4*)(Wh + s * HF + h * F + f4);
    float* o = out + d * HF + h * F + f4;
    atomicAdd(o + 0, att * w.x);
    atomicAdd(o + 1, att * w.y);
    atomicAdd(o + 2, att * w.z);
    atomicAdd(o + 3, att * w.w);
}

// ------------------------------------------------------------------
extern "C" void kernel_launch(void* const* d_in, const int* in_sizes, int n_in,
                              void* d_out, int out_size, void* d_ws, size_t ws_size,
                              hipStream_t stream) {
    (void)in_sizes; (void)n_in; (void)out_size; (void)ws_size;

    const float* x  = (const float*)d_in[0];          // [N, 128] f32
    const int*   ei = (const int*)d_in[1];            // [2, E]   (harness int32)
    const float* W  = (const float*)d_in[2];          // [128, 128] f32
    const float* a  = (const float*)d_in[3];          // [4, 64] f32
    float* out = (float*)d_out;                       // [N, 128] f32

    // workspace carve-up (~40.8 MB)
    char* ws = (char*)d_ws;
    float* Wh   = (float*)ws;                                   // N*128 f32
    float* as   = (float*)(ws + (size_t)N_NODES * HF * 4);      // N*4
    float* ad   = as  + N_NODES * NH;                           // N*4
    float* den  = ad  + N_NODES * NH;                           // N*4
    float* expe = den + N_NODES * NH;                           // E*4

    const int* src = ei;
    const int* dst = ei + N_EDGES;

    gat_zero_kernel <<<(N_NODES * HF + 255) / 256, 256, 0, stream>>>(out, den);
    gat_gemm_kernel <<<N_NODES / 16,               256, 0, stream>>>(x, W, Wh);
    gat_alpha_kernel<<<(N_NODES * NH + 255) / 256, 256, 0, stream>>>(Wh, a, as, ad);
    gat_edge1_kernel<<<N_EDGES / 256,              256, 0, stream>>>(src, dst, as, ad, expe, den);
    gat_edge2_kernel<<<(N_EDGES * 32) / 256,       256, 0, stream>>>(src, dst, Wh, expe, den, out);
}
